// Backbone_Global_Local_feat_12807592477199
// MI455X (gfx1250) — matrix operand order repaired and checked by reading, not compiled, not run on hardware
//
#include <hip/hip_runtime.h>
#include <hip/hip_bf16.h>
#include <stdint.h>

// ---------------------------------------------------------------------------
// Types
// ---------------------------------------------------------------------------
typedef __attribute__((ext_vector_type(16))) _Float16 v16h;
typedef __attribute__((ext_vector_type(8)))  _Float16 v8h;
typedef __attribute__((ext_vector_type(8)))  float    v8f;
typedef unsigned int       u32;
typedef unsigned long long u64;
typedef __attribute__((ext_vector_type(4))) u32 u32x4;
typedef __attribute__((ext_vector_type(8))) int  i32x8;
typedef __attribute__((ext_vector_type(4))) int  i32x4;

#define EPI_PRELU_F16 0   // conv -> PReLU -> f16 out (feeds next conv)
#define EPI_BN_ADD    1   // conv -> BN -> + shortcut -> f32 out
#define EPI_BN        2   // conv -> BN -> f32 out (shortcut conv)
#define EPI_BIAS_RELU 3   // conv -> +bias -> ReLU -> f32 out (head convs)
#define EPI_BN_PRELU  4   // conv -> BN -> PReLU -> f32 out (stem)

#if defined(__AMDGCN__) && __has_builtin(__builtin_amdgcn_tensor_load_to_lds)
#define HAS_TDM 1
#else
#define HAS_TDM 0
#endif

// ---------------------------------------------------------------------------
// Implicit-GEMM conv via V_WMMA_F32_16X16X32_F16, double-buffered LDS.
// GEMM view: C[M=Cout][N=B*Ho*Wo] = sum_K  W[M][K] * Patch[K][N],
//            K = Cin*KS*KS (OIHW weight flattening == row-major [Cout][K]).
// Workgroup tile: 64(M) x 128(N), BK=32. 8 waves, each wave 32x32 (4 accs).
// Slab k+1 is staged (TDM for weights, VALU gather for patches) into the
// ping-pong LDS buffers while slab k feeds the WMMAs; one barrier per K-step.
// ---------------------------------------------------------------------------
template<int KS, int EPI>
__global__ __launch_bounds__(256) void k_conv_wmma(
    const _Float16* __restrict__ X, const _Float16* __restrict__ Wt,
    float* __restrict__ outF, _Float16* __restrict__ outH,
    const float* __restrict__ p0, const float* __restrict__ p1,
    const float* __restrict__ p2,
    const float* __restrict__ sc, int scH, int scW, int scStr,
    int B, int Cin, int H, int W, int Cout, int Ho, int Wo, int stride)
{
  constexpr int PAD = KS / 2;
  const int Ktot = Cin * KS * KS;
  const int HoWo = Ho * Wo;
  const int Npix = B * HoWo;
  const int tid  = threadIdx.x;
  const int lane = tid & 31, wid = tid >> 5;
  const int waveM = wid & 1, waveN = wid >> 1;        // 2 x 4 wave grid
  const int m0blk = blockIdx.y * 64;
  const int n0blk = blockIdx.x * 128;

  __shared__ __align__(16) _Float16 As[2][64][32];    // [buf][m][k]
  __shared__ __align__(16) _Float16 Bs[2][128][32];   // [buf][n][k]

  v8f acc[2][2];
#pragma unroll
  for (int mi = 0; mi < 2; ++mi)
#pragma unroll
    for (int ni = 0; ni < 2; ++ni)
#pragma unroll
      for (int v = 0; v < 8; ++v) acc[mi][ni][v] = 0.f;

  // B-gather mapping: 2 threads per n-row, 16 k-halfs each.
  const int nrow = tid >> 1;
  const int kseg = (tid & 1) * 16;
  const int nIdx = n0blk + nrow;
  const bool nOK = (nIdx < Npix);
  int bN = 0, ohN = 0, owN = 0;
  if (nOK) { bN = nIdx / HoWo; int rem = nIdx - bN * HoWo; ohN = rem / Wo; owN = rem - ohN * Wo; }
  const int ihBase = ohN * stride - PAD;
  const int iwBase = owN * stride - PAD;

  // Stage slab starting at k0 into buffer pb (no-op past the end of K).
  auto stage = [&](int k0, int pb) {
    if (k0 >= Ktot) return;
    // ---- A tile (weights) via TDM: 2D tile 32(K) x 64(M), stride Ktot ----
#if HAS_TDM
    if (wid == 0) {
      u64 ga = (u64)(uintptr_t)Wt + 2ull * ((u64)m0blk * (u64)Ktot + (u64)k0);
      u32 ldsOff = (u32)(u64)(uintptr_t)&As[pb][0][0]; // low 32 bits = LDS offset
      u32 dim0 = (u32)(Ktot - k0);                     // remaining K (OOB -> 0-fill)
      u32 dim1 = (u32)(Cout - m0blk);
      u32x4 g0;
      g0.x = 1u;                                       // count=1, user descriptor
      g0.y = ldsOff;                                   // lds_addr
      g0.z = (u32)ga;                                  // global_addr[31:0]
      g0.w = (u32)((ga >> 32) & 0x1FFFFFFu) | (2u << 30); // addr[56:32] | type=2
      i32x8 g1;
      g1[0] = (int)(1u << 16);                         // data_size = 2B
      g1[1] = (int)((dim0 & 0xFFFFu) << 16);           // tensor_dim0[15:0]
      g1[2] = (int)(((dim0 >> 16) & 0xFFFFu) | ((dim1 & 0xFFFFu) << 16));
      g1[3] = (int)(((dim1 >> 16) & 0xFFFFu) | (32u << 16)); // tile_dim0 = 32
      g1[4] = 64;                                      // tile_dim1 = 64, tile_dim2 = 0
      g1[5] = (int)(u32)Ktot;                          // tensor_dim0_stride lo
      g1[6] = 0;
      g1[7] = 0;
      i32x4 gz  = {0, 0, 0, 0};
      i32x8 gz8 = {0, 0, 0, 0, 0, 0, 0, 0};
      __builtin_amdgcn_tensor_load_to_lds(g0, g1, gz, gz, gz8, 0);
    }
#else
    {   // fallback: plain global->reg->LDS staging of the A tile
      int m = tid >> 2, seg = tid & 3;
      __align__(16) _Float16 a8[8];
#pragma unroll
      for (int q = 0; q < 8; ++q) {
        int kg = k0 + seg * 8 + q;
        a8[q] = (kg < Ktot) ? Wt[(u64)(m0blk + m) * Ktot + kg] : (_Float16)0.f;
      }
      *(v8h*)&As[pb][m][seg * 8] = *(const v8h*)a8;
    }
#endif
    // ---- B tile (im2col patches) gathered into LDS ----
    v8h lo, hi;
#pragma unroll
    for (int q = 0; q < 8; ++q) { lo[q] = (_Float16)0.f; hi[q] = (_Float16)0.f; }
    if (nOK) {
      if (k0 + 32 < Ktot) {   // hint next K-slab into GL2
        int ciN = (k0 + 32) / (KS * KS);
        int ih0 = ihBase < 0 ? 0 : ihBase;
        __builtin_prefetch(&X[(((u64)bN * Cin + ciN) * H + ih0) * W], 0, 0);
      }
#pragma unroll
      for (int kk = 0; kk < 16; ++kk) {
        int kg = k0 + kseg + kk;
        _Float16 val = (_Float16)0.f;
        if (kg < Ktot) {
          int ci, kh, kw;
          if (KS == 1) { ci = kg; kh = 0; kw = 0; }
          else { ci = kg / 9; int r = kg - ci * 9; kh = r / 3; kw = r - kh * 3; }
          int ih = ihBase + kh, iw = iwBase + kw;
          if (ih >= 0 && ih < H && iw >= 0 && iw < W)
            val = X[(((u64)bN * Cin + ci) * H + ih) * W + iw];
        }
        if (kk < 8) lo[kk] = val; else hi[kk - 8] = val;
      }
    }
    *(v8h*)&Bs[pb][nrow][kseg]     = lo;
    *(v8h*)&Bs[pb][nrow][kseg + 8] = hi;
  };

  stage(0, 0);
  int pb = 0;
  for (int k0 = 0; k0 < Ktot; k0 += 32) {
#if HAS_TDM
    if (wid == 0) __builtin_amdgcn_s_wait_tensorcnt(0);  // TDM for slab k0 done
#endif
    __syncthreads();            // slab k0 visible; buffer pb^1 free to refill
    stage(k0 + 32, pb ^ 1);     // overlap next-slab DMA/gather with WMMAs

    // ---- fragments (documented CDNA5 WMMA VGPR layouts) + 4 WMMAs ----
    {
      const int mrow = lane & 15;
      const int aK = (lane >= 16) ? 8 : 0;    // A: lanes>=16 hold K+8 pairs
      const int bK = (lane >= 16) ? 8 : 0;    // B: lanes>=16 hold K+8 pairs
      v16h aF[2], bF[2];
#pragma unroll
      for (int mi = 0; mi < 2; ++mi) {
        int m = 32 * waveM + 16 * mi + mrow;
        v8h l0 = *(const v8h*)&As[pb][m][aK];        // K = aK .. aK+7
        v8h h0 = *(const v8h*)&As[pb][m][aK + 16];   // K = aK+16 .. aK+23
        aF[mi] = __builtin_shufflevector(l0, h0, 0,1,2,3,4,5,6,7,8,9,10,11,12,13,14,15);
      }
#pragma unroll
      for (int ni = 0; ni < 2; ++ni) {
        int n = 32 * waveN + 16 * ni + mrow;
        v8h l0 = *(const v8h*)&Bs[pb][n][bK];        // K = bK .. bK+7
        v8h h0 = *(const v8h*)&Bs[pb][n][bK + 16];   // K = bK+16 .. bK+23
        bF[ni] = __builtin_shufflevector(l0, h0, 0,1,2,3,4,5,6,7,8,9,10,11,12,13,14,15);
      }
#pragma unroll
      for (int mi = 0; mi < 2; ++mi)
#pragma unroll
        for (int ni = 0; ni < 2; ++ni)
          acc[mi][ni] = __builtin_amdgcn_wmma_f32_16x16x32_f16(
              false, aF[mi], false, bF[ni], (short)0, acc[mi][ni], false, false);
    }
    pb ^= 1;
  }

  // ---- fused epilogue + store ----
  const int coBase = m0blk + 32 * waveM;
  const int nBase  = n0blk + 32 * waveN;
  const int vAdd = (lane >= 16) ? 8 : 0;
  const int nl = lane & 15;
#pragma unroll
  for (int ni = 0; ni < 2; ++ni) {
    int n = nBase + 16 * ni + nl;
    if (n >= Npix) continue;
    int b = n / HoWo; int rem = n - b * HoWo; int oh = rem / Wo; int ow = rem - oh * Wo;
#pragma unroll
    for (int mi = 0; mi < 2; ++mi) {
#pragma unroll
      for (int v = 0; v < 8; ++v) {
        int co = coBase + 16 * mi + v + vAdd;
        float w = acc[mi][ni][v];
        u64 oidx = (((u64)b * Cout + co) * Ho + oh) * Wo + ow;
        if (EPI == EPI_PRELU_F16) {
          float y = (w >= 0.f) ? w : p0[co] * w;
          outH[oidx] = (_Float16)y;
        } else if (EPI == EPI_BN_ADD) {
          float y = w * p0[co] + p1[co]
                  + sc[(((u64)b * Cout + co) * scH + (u64)oh * scStr) * scW + (u64)ow * scStr];
          outF[oidx] = y;
        } else if (EPI == EPI_BN) {
          outF[oidx] = w * p0[co] + p1[co];
        } else if (EPI == EPI_BIAS_RELU) {
          float y = w + p0[co];
          outF[oidx] = y > 0.f ? y : 0.f;
        } else {  // EPI_BN_PRELU
          float y = w * p0[co] + p1[co];
          outF[oidx] = (y >= 0.f) ? y : p2[co] * y;
        }
      }
    }
  }
}

// ---------------------------------------------------------------------------
// Elementwise helpers
// ---------------------------------------------------------------------------
__global__ void k_bn_prep(const float* __restrict__ g, const float* __restrict__ b,
                          const float* __restrict__ m, const float* __restrict__ v,
                          float* __restrict__ s, float* __restrict__ t, int C) {
  int i = blockIdx.x * blockDim.x + threadIdx.x;
  if (i >= C) return;
  float sv = g[i] * rsqrtf(v[i] + 1e-5f);
  s[i] = sv;
  t[i] = b[i] - m[i] * sv;
}

__global__ void k_bn_cvt(const float* __restrict__ x, _Float16* __restrict__ y,
                         const float* __restrict__ s, const float* __restrict__ t,
                         int C, int HW, long n) {
  long i = (long)blockIdx.x * blockDim.x + threadIdx.x;
  if (i >= n) return;
  float v = x[i];
  if (s) { int c = (int)((i / HW) % C); v = v * s[c] + t[c]; }
  y[i] = (_Float16)v;
}

__global__ void k_cvt_f16(const float* __restrict__ x, _Float16* __restrict__ y, long n) {
  long i = (long)blockIdx.x * blockDim.x + threadIdx.x;
  if (i >= n) return;
  y[i] = (_Float16)x[i];
}

__global__ void k_crop_gather(const float* __restrict__ f2, const int* __restrict__ locs,
                              float* __restrict__ out, int B, int C, int H, int W, int ci) {
  int i = blockIdx.x * blockDim.x + threadIdx.x;
  int total = B * C * 49;
  if (i >= total) return;
  int b = i / (C * 49); int r = i - b * C * 49; int c = r / 49;
  int p = r - c * 49; int dy = p / 7, dx = p - dy * 7;
  int l0 = locs[(b * 5 + ci) * 2 + 0];
  int l1 = locs[(b * 5 + ci) * 2 + 1];
  int hh = l1 + dy - 3, ww = l0 + dx - 3;   // HALF = 3 padding
  float v = 0.f;
  if (hh >= 0 && hh < H && ww >= 0 && ww < W)
    v = f2[(((long)b * C + c) * H + hh) * W + ww];
  out[i] = v;
}

__global__ void k_pool(const float* __restrict__ x, float* __restrict__ out,
                       int B, int C, int HW, int ldOut, int off) {
  int i = blockIdx.x * blockDim.x + threadIdx.x;
  if (i >= B * C) return;
  int b = i / C, c = i - b * C;
  const float* p = x + ((long)b * C + c) * HW;
  float s = 0.f;
  for (int j = 0; j < HW; ++j) s += p[j];
  out[(long)b * ldOut + off + c] = s / (float)HW;
}

// ---------------------------------------------------------------------------
// Host side
// ---------------------------------------------------------------------------
struct BNp { const float *b, *g, *m, *v; };
struct Unit { const float* a; BNp bn1, bn2, scbn;
              const float *scw, *w1, *w2; int cin, cout, stride; };

static void conv_run(int KS, int EPI,
                     const _Float16* X, const _Float16* Wt,
                     float* oF, _Float16* oH,
                     const float* p0, const float* p1, const float* p2,
                     const float* sc, int scH, int scW, int scStr,
                     int B, int Cin, int H, int W, int Cout, int stride,
                     hipStream_t st, int* pHo, int* pWo) {
  int pad = KS / 2;
  int Ho = (H + 2 * pad - KS) / stride + 1;
  int Wo = (W + 2 * pad - KS) / stride + 1;
  if (pHo) *pHo = Ho;
  if (pWo) *pWo = Wo;
  int Npix = B * Ho * Wo;
  dim3 g((Npix + 127) / 128, Cout / 64), blk(256);
#define LNCH(ks, ep) k_conv_wmma<ks, ep><<<g, blk, 0, st>>>( \
      X, Wt, oF, oH, p0, p1, p2, sc, scH, scW, scStr, B, Cin, H, W, Cout, Ho, Wo, stride)
  if (KS == 3 && EPI == EPI_PRELU_F16)      LNCH(3, EPI_PRELU_F16);
  else if (KS == 3 && EPI == EPI_BN_ADD)    LNCH(3, EPI_BN_ADD);
  else if (KS == 3 && EPI == EPI_BIAS_RELU) LNCH(3, EPI_BIAS_RELU);
  else if (KS == 3 && EPI == EPI_BN_PRELU)  LNCH(3, EPI_BN_PRELU);
  else                                      LNCH(1, EPI_BN);
#undef LNCH
}

extern "C" void kernel_launch(void* const* d_in, const int* in_sizes, int n_in,
                              void* d_out_v, int out_size, void* d_ws, size_t ws_size,
                              hipStream_t stream) {
  (void)in_sizes; (void)n_in; (void)out_size;
  const int B = 32;
  const float* imgs = (const float*)d_in[0];
  const int*   locs = (const int*)d_in[1];
  float* d_out = (float*)d_out_v;

  // ---- parse params in JAX pytree leaf order (dict keys sorted, None dropped)
  int li = 2;
  auto nxt = [&]() -> const float* { return (const float*)d_in[li++]; };
  auto parseBN = [&]() { BNp r; r.b = nxt(); r.g = nxt(); r.m = nxt(); r.v = nxt(); return r; };
  auto parseUnit = [&](int ci, int co, int s) {
    Unit u; u.cin = ci; u.cout = co; u.stride = s;
    u.a = nxt(); u.bn1 = parseBN(); u.bn2 = parseBN();
    if (ci != co) { u.scbn = parseBN(); u.scw = nxt(); }
    else { u.scw = nullptr; u.scbn = BNp{nullptr, nullptr, nullptr, nullptr}; }
    u.w1 = nxt(); u.w2 = nxt();
    return u;
  };

  struct CropBranch { const float* bias; Unit units[4]; const float* w; };
  CropBranch crops[5];
  const int cropSpec[4][3] = {{128,256,2},{256,256,1},{256,512,2},{512,512,1}};
  for (int i = 0; i < 5; ++i) {                         // 'crops'
    crops[i].bias = nxt();
    for (int uI = 0; uI < 4; ++uI)
      crops[i].units[uI] = parseUnit(cropSpec[uI][0], cropSpec[uI][1], cropSpec[uI][2]);
    crops[i].w = nxt();
  }
  const float* stemA = nxt(); BNp stemBN = parseBN();   // 'input'
  const float* stemW = nxt();
  Unit layers[24]; int nU = 0;                          // 'layers'
  const int blkIn[4] = {64, 64, 128, 256};
  const int blkD[4]  = {64, 128, 256, 512};
  const int blkN[4]  = {3, 4, 14, 3};
  for (int bI = 0; bI < 4; ++bI) {
    layers[nU++] = parseUnit(blkIn[bI], blkD[bI], 2);
    for (int j = 1; j < blkN[bI]; ++j) layers[nU++] = parseUnit(blkD[bI], blkD[bI], 1);
  }
  const float* outB = nxt(); const float* outW = nxt(); // 'out'

  // ---- workspace (bump allocator) ----
  size_t off = 0;
  auto alloc = [&](size_t bytes) {
    void* p = (char*)d_ws + off;
    off = (off + bytes + 255) & ~(size_t)255;
    return p;
  };
  const size_t MAXACT = (size_t)B * 64 * 112 * 112;     // 25.69M elems
  float*    A0 = (float*)alloc(MAXACT * 4);
  float*    A1 = (float*)alloc(MAXACT * 4);
  _Float16* T0 = (_Float16*)alloc(MAXACT * 2);
  _Float16* R0 = (_Float16*)alloc(MAXACT * 2);
  float*    S  = (float*)alloc((size_t)B * 128 * 28 * 28 * 4);
  _Float16* WB = (_Float16*)alloc((size_t)512 * 512 * 9 * 2);
  float*    F2 = (float*)alloc((size_t)B * 128 * 28 * 28 * 4);
  float*    CR = (float*)alloc((size_t)B * 128 * 49 * 4);
  float* bnS1 = (float*)alloc(512 * 4); float* bnT1 = (float*)alloc(512 * 4);
  float* bnS2 = (float*)alloc(512 * 4); float* bnT2 = (float*)alloc(512 * 4);
  float* bnS3 = (float*)alloc(512 * 4); float* bnT3 = (float*)alloc(512 * 4);
  if (off > ws_size) return;   // workspace too small; nothing we can do

  auto gridFor = [](long n) { return dim3((unsigned)((n + 255) / 256)); };

  // ---- residual unit: bn1 -> conv1(+PReLU,f16) -> conv2(+bn2 + shortcut) ----
  auto unit_run = [&](const Unit& u, const float* xin, float* xout,
                      int H, int W, int* pHo, int* pWo) {
    long nIn = (long)B * u.cin * H * W;
    k_bn_prep<<<(u.cin + 63) / 64, 64, 0, stream>>>(u.bn1.g, u.bn1.b, u.bn1.m, u.bn1.v, bnS1, bnT1, u.cin);
    k_bn_prep<<<(u.cout + 63) / 64, 64, 0, stream>>>(u.bn2.g, u.bn2.b, u.bn2.m, u.bn2.v, bnS2, bnT2, u.cout);
    k_bn_cvt<<<gridFor(nIn), 256, 0, stream>>>(xin, T0, bnS1, bnT1, u.cin, H * W, nIn);
    long nW1 = (long)u.cout * u.cin * 9;
    k_cvt_f16<<<gridFor(nW1), 256, 0, stream>>>(u.w1, WB, nW1);
    int h1, w1;
    conv_run(3, EPI_PRELU_F16, T0, WB, nullptr, R0, u.a, nullptr, nullptr,
             nullptr, 0, 0, 0, B, u.cin, H, W, u.cout, 1, stream, &h1, &w1);
    const float* scP; int scH, scW, scStr;
    int Ho, Wo;
    if (u.scw) {
      k_bn_prep<<<(u.cout + 63) / 64, 64, 0, stream>>>(u.scbn.g, u.scbn.b, u.scbn.m, u.scbn.v, bnS3, bnT3, u.cout);
      k_bn_cvt<<<gridFor(nIn), 256, 0, stream>>>(xin, T0, nullptr, nullptr, u.cin, H * W, nIn);
      long nWs = (long)u.cout * u.cin;
      k_cvt_f16<<<gridFor(nWs), 256, 0, stream>>>(u.scw, WB, nWs);
      conv_run(1, EPI_BN, T0, WB, S, nullptr, bnS3, bnT3, nullptr,
               nullptr, 0, 0, 0, B, u.cin, H, W, u.cout, u.stride, stream, &Ho, &Wo);
      scP = S; scH = Ho; scW = Wo; scStr = 1;
    } else {
      scP = xin; scH = H; scW = W; scStr = u.stride;   // identity x[:, :, ::s, ::s]
    }
    long nW2 = (long)u.cout * u.cout * 9;
    k_cvt_f16<<<gridFor(nW2), 256, 0, stream>>>(u.w2, WB, nW2);
    conv_run(3, EPI_BN_ADD, R0, WB, xout, nullptr, bnS2, bnT2, nullptr,
             scP, scH, scW, scStr, B, u.cout, H, W, u.cout, u.stride, stream, &Ho, &Wo);
    *pHo = Ho; *pWo = Wo;
  };

  // ---- stem: prelu(bn(conv(imgs))) ----
  long nImg = (long)B * 3 * 112 * 112;
  k_bn_cvt<<<gridFor(nImg), 256, 0, stream>>>(imgs, T0, nullptr, nullptr, 3, 112 * 112, nImg);
  k_cvt_f16<<<gridFor(64 * 27), 256, 0, stream>>>(stemW, WB, 64 * 27);
  k_bn_prep<<<1, 64, 0, stream>>>(stemBN.g, stemBN.b, stemBN.m, stemBN.v, bnS1, bnT1, 64);
  int Ho, Wo;
  conv_run(3, EPI_BN_PRELU, T0, WB, A0, nullptr, bnS1, bnT1, stemA,
           nullptr, 0, 0, 0, B, 3, 112, 112, 64, 1, stream, &Ho, &Wo);

  // ---- backbone blocks ----
  float* cur = A0; float* alt = A1; int H = 112, W = 112; int uidx = 0;
  for (int bI = 0; bI < 4; ++bI) {
    for (int j = 0; j < blkN[bI]; ++j) {
      unit_run(layers[uidx++], cur, alt, H, W, &Ho, &Wo);
      float* tp = cur; cur = alt; alt = tp; H = Ho; W = Wo;
    }
    if (bI == 1)   // save f2 = feats[1] (B,128,28,28) for the crop branches
      (void)hipMemcpyAsync(F2, cur, (size_t)B * 128 * 28 * 28 * 4,
                           hipMemcpyDeviceToDevice, stream);
  }

  // ---- global head: relu(conv(f4) + b) -> mean -> d_out[:, 0:64] ----
  long nF4 = (long)B * 512 * 7 * 7;
  k_bn_cvt<<<gridFor(nF4), 256, 0, stream>>>(cur, T0, nullptr, nullptr, 512, 49, nF4);
  k_cvt_f16<<<gridFor((long)64 * 512 * 9), 256, 0, stream>>>(outW, WB, (long)64 * 512 * 9);
  conv_run(3, EPI_BIAS_RELU, T0, WB, S, nullptr, outB, nullptr, nullptr,
           nullptr, 0, 0, 0, B, 512, 7, 7, 64, 1, stream, &Ho, &Wo);
  k_pool<<<gridFor((long)B * 64), 256, 0, stream>>>(S, d_out, B, 64, 49, 384, 0);

  // ---- 5 local crop branches -> d_out[:, 64 + 64*i : ] ----
  for (int ci = 0; ci < 5; ++ci) {
    k_crop_gather<<<gridFor((long)B * 128 * 49), 256, 0, stream>>>(F2, locs, CR, B, 128, 28, 28, ci);
    const float* curc = CR; int h = 7, w = 7; int pi = 0;
    float* pp[2] = {A0, A1};
    for (int uI = 0; uI < 4; ++uI) {
      float* outc = pp[pi]; pi ^= 1;
      int ho, wo;
      unit_run(crops[ci].units[uI], curc, outc, h, w, &ho, &wo);
      curc = outc; h = ho; w = wo;
    }
    long nY = (long)B * 512 * h * w;
    k_bn_cvt<<<gridFor(nY), 256, 0, stream>>>(curc, T0, nullptr, nullptr, 512, h * w, nY);
    k_cvt_f16<<<gridFor((long)64 * 512 * 9), 256, 0, stream>>>(crops[ci].w, WB, (long)64 * 512 * 9);
    int ho2, wo2;
    conv_run(3, EPI_BIAS_RELU, T0, WB, S, nullptr, crops[ci].bias, nullptr, nullptr,
             nullptr, 0, 0, 0, B, 512, h, w, 64, 1, stream, &ho2, &wo2);
    k_pool<<<gridFor((long)B * 64), 256, 0, stream>>>(S, d_out, B, 64, ho2 * wo2, 384, 64 + 64 * ci);
  }
}
